// GCNBlock_6820408066453
// MI455X (gfx1250) — matrix operand order repaired and checked by reading, not compiled, not run on hardware
//
#include <hip/hip_runtime.h>
#include <hip/hip_bf16.h>

// Problem constants from the reference
#define BB 8
#define NN 10000
#define DD 64
#define EE 160000
#define ROWS (BB * NN)            // 80000 flattened (b,n) rows
#define SLAB ((size_t)NN * DD)    // per-batch slab in elements

typedef __attribute__((ext_vector_type(16))) __bf16          v16bf;
typedef __attribute__((ext_vector_type(8)))  float           v8f;
typedef __attribute__((ext_vector_type(8)))  unsigned int    v8ui;

// ---------------------------------------------------------------------------
// Convert W0/W1/W2 (fp32, row-major 64x64) to bf16 into workspace.
// Native __bf16 cast -> v_cvt_pk_bf16_f32 on gfx1250.
// ---------------------------------------------------------------------------
__global__ __launch_bounds__(256) void convw_kernel(
    const float* __restrict__ W0, const float* __restrict__ W1,
    const float* __restrict__ W2, __bf16* __restrict__ Wbf) {
    int i = blockIdx.x * 256 + threadIdx.x;        // 0 .. 3*4096-1
    const float* src = (i < 4096) ? W0 : (i < 8192 ? W1 : W2);
    Wbf[i] = (__bf16)src[i & 4095];
}

// ---------------------------------------------------------------------------
// Y[row, f] = relu( sum_d X[row, d] * W[f, d] ),  row-major, D = 64.
// One wave computes a 16x64 output tile via 8x v_wmma_f32_16x16x32_bf16.
// ---------------------------------------------------------------------------
__device__ __forceinline__ v16bf load_a_frag(const float* __restrict__ xr,
                                             int kh, int half) {
    // A 16x32 layout (matching B): lane holds K = kh*32 + half*16 + {0..15}
    // in elements 0..15.
    const int k0 = kh * 32 + half * 16;
    const int k1 = k0 + 8;
    const float4 f0 = *(const float4*)(xr + k0);
    const float4 f1 = *(const float4*)(xr + k0 + 4);
    const float4 f2 = *(const float4*)(xr + k1);
    const float4 f3 = *(const float4*)(xr + k1 + 4);
    v16bf t;
    t[0]  = (__bf16)f0.x;  t[1]  = (__bf16)f0.y;
    t[2]  = (__bf16)f0.z;  t[3]  = (__bf16)f0.w;
    t[4]  = (__bf16)f1.x;  t[5]  = (__bf16)f1.y;
    t[6]  = (__bf16)f1.z;  t[7]  = (__bf16)f1.w;
    t[8]  = (__bf16)f2.x;  t[9]  = (__bf16)f2.y;
    t[10] = (__bf16)f2.z;  t[11] = (__bf16)f2.w;
    t[12] = (__bf16)f3.x;  t[13] = (__bf16)f3.y;
    t[14] = (__bf16)f3.z;  t[15] = (__bf16)f3.w;
    return t;
}

__device__ __forceinline__ v16bf load_b_frag(const __bf16* __restrict__ wbf,
                                             int ct, int kh, int nn, int khalf) {
    // B 32x16 fragment in contiguous k order (same as A): lane (col = nn, K-half = khalf) holds 16
    // consecutive K values -> 16 contiguous bf16 of row f of W (B[k,n]=W[f,d]).
    const int f  = ct * 16 + nn;
    const int d0 = kh * 32 + khalf * 16;
    v8ui raw = *(const v8ui*)(wbf + f * 64 + d0);   // 32B aligned load
    return __builtin_bit_cast(v16bf, raw);
}

__global__ __launch_bounds__(256) void gemm_relu_kernel(
    const float* __restrict__ X, const __bf16* __restrict__ Wbf,
    float* __restrict__ Y) {
    const int lane = threadIdx.x & 31;
    const int wave = threadIdx.x >> 5;
    const int tile = blockIdx.x * 8 + wave;   // 5000 tiles total, exact fit
    const int row0 = tile * 16;
    const int m    = lane & 15;               // A row-in-tile == B/C column lane
    const int half = lane >> 4;

    v8f acc0 = {}, acc1 = {}, acc2 = {}, acc3 = {};
    const float* xr = X + (size_t)(row0 + m) * DD;

#pragma unroll
    for (int kh = 0; kh < 2; ++kh) {
        v16bf a  = load_a_frag(xr, kh, half);
        v16bf b0 = load_b_frag(Wbf, 0, kh, m, half);
        v16bf b1 = load_b_frag(Wbf, 1, kh, m, half);
        v16bf b2 = load_b_frag(Wbf, 2, kh, m, half);
        v16bf b3 = load_b_frag(Wbf, 3, kh, m, half);
        acc0 = __builtin_amdgcn_wmma_f32_16x16x32_bf16(false, a, false, b0,
                                                       (short)0, acc0, false, false);
        acc1 = __builtin_amdgcn_wmma_f32_16x16x32_bf16(false, a, false, b1,
                                                       (short)0, acc1, false, false);
        acc2 = __builtin_amdgcn_wmma_f32_16x16x32_bf16(false, a, false, b2,
                                                       (short)0, acc2, false, false);
        acc3 = __builtin_amdgcn_wmma_f32_16x16x32_bf16(false, a, false, b3,
                                                       (short)0, acc3, false, false);
    }

    // C/D layout: VGPR v holds (row = v + 8*half, col = lane&15)
#pragma unroll
    for (int v = 0; v < 8; ++v) {
        float* yr = Y + (size_t)(row0 + v + half * 8) * DD + m;
        yr[0]  = fmaxf(acc0[v], 0.0f);
        yr[16] = fmaxf(acc1[v], 0.0f);
        yr[32] = fmaxf(acc2[v], 0.0f);
        yr[48] = fmaxf(acc3[v], 0.0f);
    }
}

// ---------------------------------------------------------------------------
// Zero-fill (accumulation targets for the scatter-add).
// ---------------------------------------------------------------------------
__global__ __launch_bounds__(256) void fill_zero_kernel(float4* __restrict__ p, int n4) {
    int i = blockIdx.x * 256 + threadIdx.x;
    if (i < n4) p[i] = make_float4(0.f, 0.f, 0.f, 0.f);
}

// ---------------------------------------------------------------------------
// Q[b, r, :] += vals[e] * T[b, c, :]  for every edge e, every batch b.
// 64 lanes cover D (coalesced 256B row gather from L2); fp32 global atomics
// do the scatter.  4 edges per 256-thread block.
// ---------------------------------------------------------------------------
__global__ __launch_bounds__(256) void spmm_kernel(
    const float* __restrict__ T, const float* __restrict__ vals,
    const int* __restrict__ rows, const int* __restrict__ cols,
    float* __restrict__ Q) {
    const int d  = threadIdx.x & 63;
    const int el = threadIdx.x >> 6;
    const int e  = blockIdx.x * 4 + el;
    if (e >= EE) return;

    const int   r = rows[e];
    const int   c = cols[e];
    const float v = vals[e];

    const float* src = T + (size_t)c * DD + d;
    float*       dst = Q + (size_t)r * DD + d;

#pragma unroll
    for (int b = 0; b < BB; ++b) {
        float x = src[(size_t)b * SLAB];
        atomicAdd(dst + (size_t)b * SLAB, v * x);
    }
}

// ---------------------------------------------------------------------------
extern "C" void kernel_launch(void* const* d_in, const int* in_sizes, int n_in,
                              void* d_out, int out_size, void* d_ws, size_t ws_size,
                              hipStream_t stream) {
    const float* x    = (const float*)d_in[0];
    const float* W0   = (const float*)d_in[1];
    const float* W1   = (const float*)d_in[2];
    const float* W2   = (const float*)d_in[3];
    const float* avls = (const float*)d_in[4];
    const int*   arow = (const int*)d_in[5];
    const int*   acol = (const int*)d_in[6];
    float*       out  = (float*)d_out;

    const size_t S = (size_t)BB * NN * DD * sizeof(float);   // 20.48 MB
    char* ws = (char*)d_ws;
    float*   T   = (float*)ws;            // GEMM output
    float*   Q   = (float*)(ws + S);      // SpMM output (ping-pong)
    __bf16*  Wbf = (__bf16*)(ws + 2 * S);

    const int n4        = (BB * NN * DD) / 4;    // 1,280,000 float4s
    const int zero_grid = (n4 + 255) / 256;      // 5000
    const int gemm_grid = ROWS / (16 * 8);       // 625, exact
    const int spmm_grid = EE / 4;                // 40000, exact

    // W -> bf16 once
    convw_kernel<<<48, 256, 0, stream>>>(W0, W1, W2, Wbf);

    // Layer 1: T = relu(x W0^T); Q = A T
    gemm_relu_kernel<<<gemm_grid, 256, 0, stream>>>(x, Wbf + 0 * 4096, T);
    fill_zero_kernel<<<zero_grid, 256, 0, stream>>>((float4*)Q, n4);
    spmm_kernel<<<spmm_grid, 256, 0, stream>>>(T, avls, arow, acol, Q);

    // Layer 2: T = relu(Q W1^T); Q = A T   (Q consumed by gemm before re-zero)
    gemm_relu_kernel<<<gemm_grid, 256, 0, stream>>>(Q, Wbf + 1 * 4096, T);
    fill_zero_kernel<<<zero_grid, 256, 0, stream>>>((float4*)Q, n4);
    spmm_kernel<<<spmm_grid, 256, 0, stream>>>(T, avls, arow, acol, Q);

    // Layer 3: T = relu(Q W2^T); out = A T
    gemm_relu_kernel<<<gemm_grid, 256, 0, stream>>>(Q, Wbf + 2 * 4096, T);
    fill_zero_kernel<<<zero_grid, 256, 0, stream>>>((float4*)out, n4);
    spmm_kernel<<<spmm_grid, 256, 0, stream>>>(T, avls, arow, acol, out);
}
